// multihead_attention_37340445672066
// MI455X (gfx1250) — matrix operand order repaired and checked by reading, not compiled, not run on hardware
//
#include <hip/hip_runtime.h>
#include <stdint.h>

// ---------------- problem constants ----------------
#define BATCH 4
#define S_LEN 2048
#define DM    512
#define NH    8
#define DH    64
#define NEGV  (-1e9f)

// ---------------- WMMA vector types ----------------
typedef __attribute__((ext_vector_type(16))) __bf16 v16bf;
typedef __attribute__((ext_vector_type(8)))  __bf16 v8bf;
typedef __attribute__((ext_vector_type(4)))  __bf16 v4bf;
typedef __attribute__((ext_vector_type(2)))  __bf16 v2bf;
typedef __attribute__((ext_vector_type(8)))  float  v8f;

// ---- native f32 -> bf16 conversion (RNE) ----
static __device__ __forceinline__ __bf16 bf16c(float f) { return (__bf16)f; }

// Convert 8 floats -> 8 bf16 via packed-pair converts (v_cvt_pk_bf16_f32).
static __device__ __forceinline__ v8bf cvt8(v8f x) {
#if __has_builtin(__builtin_amdgcn_cvt_pk_bf16_f32)
  v2bf p0 = __builtin_amdgcn_cvt_pk_bf16_f32(x[0], x[1]);
  v2bf p1 = __builtin_amdgcn_cvt_pk_bf16_f32(x[2], x[3]);
  v2bf p2 = __builtin_amdgcn_cvt_pk_bf16_f32(x[4], x[5]);
  v2bf p3 = __builtin_amdgcn_cvt_pk_bf16_f32(x[6], x[7]);
  v4bf a = __builtin_shufflevector(p0, p1, 0, 1, 2, 3);
  v4bf b = __builtin_shufflevector(p2, p3, 0, 1, 2, 3);
  return __builtin_shufflevector(a, b, 0, 1, 2, 3, 4, 5, 6, 7);
#else
  v8bf r;
#pragma unroll
  for (int i = 0; i < 8; ++i) r[i] = (__bf16)x[i];
  return r;
#endif
}

static __device__ __forceinline__ v16bf cat8(v8bf lo, v8bf hi) {
  return __builtin_shufflevector(lo, hi, 0,1,2,3,4,5,6,7,8,9,10,11,12,13,14,15);
}

static __device__ __forceinline__ v8f wmma_bf16(v16bf a, v16bf b, v8f c) {
  // D = A(16x32 bf16) x B(32x16 bf16) + C(16x16 f32)
  return __builtin_amdgcn_wmma_f32_16x16x32_bf16(false, a, false, b, (short)0, c,
                                                 false, false);
}

// ---- raw (pre-conversion) fragment carriers for software pipelining ----
struct RawAF { v8f lo, hi; };       // A fragment source data, f32
struct RawB  { v8bf lo, hi; };      // A- or B-fragment data already bf16

// A fragment source from row-major f32 matrix (two 32B runs per lane).
static __device__ __forceinline__ RawAF raw_a_f32(const float* A, int lda,
                                                  int row0, int k0, int lane) {
  int m = lane & 15, h = lane >> 4;
  const float* base = A + (size_t)(row0 + m) * lda + k0;
  RawAF r;
  r.lo = *(const v8f*)(base + 8 * h);
  r.hi = *(const v8f*)(base + 16 + 8 * h);
  return r;
}
static __device__ __forceinline__ v16bf a_from_f32(RawAF r) {
  return cat8(cvt8(r.lo), cvt8(r.hi));
}

// A fragment from row-major bf16 matrix (two 16B runs per lane).
static __device__ __forceinline__ RawB raw_a_bf16(const __bf16* A, int lda,
                                                  int row0, int k0, int lane) {
  int m = lane & 15, h = lane >> 4;
  const __bf16* base = A + (size_t)(row0 + m) * lda + k0;
  RawB r;
  r.lo = *(const v8bf*)(base + 8 * h);
  r.hi = *(const v8bf*)(base + 16 + 8 * h);
  return r;
}

// B fragment (32x16) from an N-major bf16 matrix Bt[n][k]; two 16B runs per lane.
static __device__ __forceinline__ RawB raw_bt(const __bf16* Bt, size_t ldb,
                                              int col0, int k0, int lane) {
  int n = lane & 15, h = lane >> 4;
  const __bf16* base = Bt + (size_t)(col0 + n) * ldb + k0;
  RawB r;
  r.lo = *(const v8bf*)(base + 8 * h);
  r.hi = *(const v8bf*)(base + 16 + 8 * h);
  return r;
}
static __device__ __forceinline__ v16bf frag(RawB r) { return cat8(r.lo, r.hi); }

// ---------------- kernel 1: weight transpose + bf16 convert ----------------
__global__ void wtrans_kernel(const float* __restrict__ W, __bf16* __restrict__ Wt) {
  int idx = blockIdx.x * blockDim.x + threadIdx.x;
  int n = idx & (DM - 1);
  int k = idx >> 9;
  Wt[(size_t)n * DM + k] = bf16c(W[(size_t)k * DM + n]);
}

// ---------------- kernel 2: QKV projection (software-pipelined) ----------------
// mode 0: dst bf16 head-split [b,h,s,d] ; mode 1: transposed [b,h,d,s]
__global__ __launch_bounds__(128) void proj_kernel(const float* __restrict__ A,
                                                   const __bf16* __restrict__ Wt,
                                                   const float* __restrict__ bias,
                                                   __bf16* __restrict__ dst,
                                                   int mode) {
  int lane = threadIdx.x & 31;
  int wid  = threadIdx.x >> 5;
  int tile = blockIdx.x * 4 + wid;          // 512 row-tiles x 32 col-tiles
  int tm = tile >> 5;
  int tn = tile & 31;
  int row0 = tm * 16, col0 = tn * 16;

  v8f c = {};
  RawAF ar = raw_a_f32(A, DM, row0, 0, lane);
  RawB  br = raw_bt(Wt, DM, col0, 0, lane);
#pragma unroll
  for (int k0 = 0; k0 < DM; k0 += 32) {
    v16bf af = a_from_f32(ar);
    v16bf bf = frag(br);
    if (k0 + 32 < DM) {                     // issue next loads before this WMMA
      ar = raw_a_f32(A, DM, row0, k0 + 32, lane);
      br = raw_bt(Wt, DM, col0, k0 + 32, lane);
    }
    c = wmma_bf16(af, bf, c);
  }

  int half = lane >> 4, n = lane & 15;
  int col = col0 + n;
  int h = col >> 6, d = col & (DH - 1);
  float bv = bias[col];
#pragma unroll
  for (int r = 0; r < 8; ++r) {
    int m  = row0 + r + 8 * half;           // global row in [0, B*S)
    int b_ = m >> 11;
    int s  = m & (S_LEN - 1);
    __bf16 val = bf16c(c[r] + bv);
    if (mode == 0)
      dst[((size_t)(b_ * NH + h) * S_LEN + s) * DH + d] = val;
    else
      dst[((size_t)(b_ * NH + h) * DH + d) * S_LEN + s] = val;
  }
}

// ---------------- kernel 3: fused scores + softmax + PV ----------------
// One block = 16 query rows of one (b,h). 128 threads = 4 waves.
// LDS: stripe[16][2048] f32 (128KB) + reduce scratch. Probs hit HBM exactly once.
__global__ __launch_bounds__(128) void attn_kernel(const __bf16* __restrict__ qh,
                                                   const __bf16* __restrict__ kh,
                                                   const __bf16* __restrict__ vt,
                                                   const unsigned char* __restrict__ mask,
                                                   float* __restrict__ probs,
                                                   __bf16* __restrict__ ctx) {
  extern __shared__ __align__(32) float sm[];
  float* stripe  = sm;                       // 16*2048
  float* red     = sm + 16 * S_LEN;          // 16*8 partials
  float* rowstat = red + 16 * 8;             // [0:16) rowmax, [16:32) rowinv

  int tid  = threadIdx.x;
  int lane = tid & 31;
  int wid  = tid >> 5;
  int half = lane >> 4, n = lane & 15;

  int bh = blockIdx.y;                       // b*NH + h
  int b  = bh >> 3;
  int h  = bh & (NH - 1);
  int q0 = blockIdx.x * 16;

  const __bf16* qp = qh + (size_t)bh * S_LEN * DH;   // [S][DH]
  const __bf16* kp = kh + (size_t)bh * S_LEN * DH;   // [S][DH]
  const __bf16* vp = vt + (size_t)bh * DH * S_LEN;   // [DH][S]
  const unsigned char* mp = mask + (size_t)b * S_LEN * S_LEN;

  // Preload the two A fragments for these 16 q-rows (Dh = 64 = 2 x K32).
  v16bf a0 = frag(raw_a_bf16(qp, DH, q0, 0,  lane));
  v16bf a1 = frag(raw_a_bf16(qp, DH, q0, 32, lane));

  // ---- Phase 1: scaled, masked scores -> LDS stripe (K frags pipelined) ----
  RawB b0r = raw_bt(kp, DH, wid * 16, 0,  lane);
  RawB b1r = raw_bt(kp, DH, wid * 16, 32, lane);
  for (int tk = wid; tk < S_LEN / 16; tk += 4) {            // wave-uniform
    int col0 = tk * 16;
    v16bf b0 = frag(b0r);
    v16bf b1 = frag(b1r);
    if (tk + 4 < S_LEN / 16) {              // prefetch next K tile
      b0r = raw_bt(kp, DH, col0 + 64, 0,  lane);
      b1r = raw_bt(kp, DH, col0 + 64, 32, lane);
    }
    v8f c = {};
    c = wmma_bf16(a0, b0, c);
    c = wmma_bf16(a1, b1, c);
    int kc = col0 + n;
#pragma unroll
    for (int r = 0; r < 8; ++r) {
      int qr = r + 8 * half;                 // stripe row 0..15
      float s = c[r] * 0.125f;               // 1/sqrt(64)
      if (mp[(size_t)(q0 + qr) * S_LEN + kc]) s = NEGV;
      stripe[qr * S_LEN + kc] = s;
    }
  }
  __syncthreads();

  // ---- Phase 2: row softmax in LDS ----
  int row = tid >> 3, chunk = tid & 7;       // 16 rows x 8 chunks x 256 elems
  float* rp = stripe + row * S_LEN + chunk * 256;
  float mx = -3.4e38f;
#pragma unroll 8
  for (int i = 0; i < 256; ++i) mx = fmaxf(mx, rp[i]);
  red[row * 8 + chunk] = mx;
  __syncthreads();
  if (chunk == 0) {
    float m2 = red[row * 8];
#pragma unroll
    for (int i = 1; i < 8; ++i) m2 = fmaxf(m2, red[row * 8 + i]);
    rowstat[row] = m2;
  }
  __syncthreads();
  float rmax = rowstat[row];
  float sum = 0.f;
#pragma unroll 8
  for (int i = 0; i < 256; ++i) { float e = __expf(rp[i] - rmax); rp[i] = e; sum += e; }
  red[row * 8 + chunk] = sum;
  __syncthreads();
  if (chunk == 0) {
    float s2 = 0.f;
#pragma unroll
    for (int i = 0; i < 8; ++i) s2 += red[row * 8 + i];
    rowstat[16 + row] = 1.0f / s2;
  }
  __syncthreads();

  // normalize in LDS + single coalesced non-temporal HBM store of the prob stripe
  float* pout = probs + ((size_t)bh * S_LEN + q0) * S_LEN;
  for (int idx = tid; idx < 16 * S_LEN; idx += 128) {
    int rr = idx >> 11;
    float p = stripe[idx] * rowstat[16 + rr];
    stripe[idx] = p;
    __builtin_nontemporal_store(p, &pout[(size_t)rr * S_LEN + (idx & (S_LEN - 1))]);
  }
  __syncthreads();

  // ---- Phase 3: context = P @ V straight from LDS (pipelined) ----
  int d0 = wid * 16;                         // 4 waves cover Dh=64
  const float* abase0 = stripe + (lane & 15) * S_LEN;
  v8f acc = {};
  RawB vr = raw_bt(vp, S_LEN, d0, 0, lane);
  v8f plo = *(const v8f*)(abase0 + 8 * half);
  v8f phi = *(const v8f*)(abase0 + 16 + 8 * half);
  for (int k0 = 0; k0 < S_LEN; k0 += 32) {
    v16bf afr = cat8(cvt8(plo), cvt8(phi));
    v16bf bfr = frag(vr);
    if (k0 + 32 < S_LEN) {                  // prefetch next P/V fragments
      const float* abase = abase0 + k0 + 32;
      plo = *(const v8f*)(abase + 8 * half);
      phi = *(const v8f*)(abase + 16 + 8 * half);
      vr  = raw_bt(vp, S_LEN, d0, k0 + 32, lane);
    }
    acc = wmma_bf16(afr, bfr, acc);
  }
#pragma unroll
  for (int r = 0; r < 8; ++r) {
    int qr = r + 8 * half;
    ctx[(size_t)(b * S_LEN + q0 + qr) * DM + h * DH + d0 + n] = bf16c(acc[r]);
  }
}

// ---------------- kernel 4: output projection (software-pipelined) ----------------
__global__ __launch_bounds__(128) void oproj_kernel(const __bf16* __restrict__ Ctx,
                                                    const __bf16* __restrict__ Wot,
                                                    const float* __restrict__ bias,
                                                    float* __restrict__ out) {
  int lane = threadIdx.x & 31;
  int wid  = threadIdx.x >> 5;
  int tile = blockIdx.x * 4 + wid;
  int tm = tile >> 5;
  int tn = tile & 31;
  int row0 = tm * 16, col0 = tn * 16;

  v8f c = {};
  RawB ar = raw_a_bf16(Ctx, DM, row0, 0, lane);
  RawB br = raw_bt(Wot, DM, col0, 0, lane);
#pragma unroll
  for (int k0 = 0; k0 < DM; k0 += 32) {
    v16bf af = frag(ar);
    v16bf bf = frag(br);
    if (k0 + 32 < DM) {
      ar = raw_a_bf16(Ctx, DM, row0, k0 + 32, lane);
      br = raw_bt(Wot, DM, col0, k0 + 32, lane);
    }
    c = wmma_bf16(af, bf, c);
  }
  int half = lane >> 4, n = lane & 15;
  int col = col0 + n;
  float bv = bias[col];
#pragma unroll
  for (int r = 0; r < 8; ++r) {
    int m = row0 + r + 8 * half;
    out[(size_t)m * DM + col] = c[r] + bv;
  }
}

// ---------------- host launcher ----------------
extern "C" void kernel_launch(void* const* d_in, const int* in_sizes, int n_in,
                              void* d_out, int out_size, void* d_ws, size_t ws_size,
                              hipStream_t stream) {
  const float*         Q    = (const float*)d_in[0];
  const float*         K    = (const float*)d_in[1];
  const float*         V    = (const float*)d_in[2];
  const unsigned char* mask = (const unsigned char*)d_in[3]; // jnp.bool_ -> 1 byte/elem
  const float*         Wq   = (const float*)d_in[4];
  const float*         bq   = (const float*)d_in[5];
  const float*         Wk   = (const float*)d_in[6];
  const float*         bk   = (const float*)d_in[7];
  const float*         Wv   = (const float*)d_in[8];
  const float*         bv   = (const float*)d_in[9];
  const float*         Wo   = (const float*)d_in[10];
  const float*         bo   = (const float*)d_in[11];

  float* out   = (float*)d_out;
  float* probs = out + (size_t)BATCH * S_LEN * DM;   // second tuple element

  // workspace carve-up
  char* ws = (char*)d_ws;
  size_t off = 0;
  __bf16* Wqt = (__bf16*)(ws + off); off += (size_t)DM * DM * 2;
  __bf16* Wkt = (__bf16*)(ws + off); off += (size_t)DM * DM * 2;
  __bf16* Wvt = (__bf16*)(ws + off); off += (size_t)DM * DM * 2;
  __bf16* Wot = (__bf16*)(ws + off); off += (size_t)DM * DM * 2;
  __bf16* qh  = (__bf16*)(ws + off); off += (size_t)BATCH * NH * S_LEN * DH * 2;
  __bf16* kh  = (__bf16*)(ws + off); off += (size_t)BATCH * NH * S_LEN * DH * 2;
  __bf16* vtb = (__bf16*)(ws + off); off += (size_t)BATCH * NH * DH * S_LEN * 2;
  __bf16* ctx = (__bf16*)(ws + off); off += (size_t)BATCH * S_LEN * DM * 2;

  // 1. weights -> bf16, N-major
  wtrans_kernel<<<DM * DM / 256, 256, 0, stream>>>(Wq, Wqt);
  wtrans_kernel<<<DM * DM / 256, 256, 0, stream>>>(Wk, Wkt);
  wtrans_kernel<<<DM * DM / 256, 256, 0, stream>>>(Wv, Wvt);
  wtrans_kernel<<<DM * DM / 256, 256, 0, stream>>>(Wo, Wot);

  // 2. projections (one 16x16 tile per wave, 4 waves/block)
  int ptiles = (BATCH * S_LEN / 16) * (DM / 16);     // 16384
  proj_kernel<<<ptiles / 4, 128, 0, stream>>>(Q, Wqt, bq, qh, 0);
  proj_kernel<<<ptiles / 4, 128, 0, stream>>>(K, Wkt, bk, kh, 0);
  proj_kernel<<<ptiles / 4, 128, 0, stream>>>(V, Wvt, bv, vtb, 1);

  // 3. fused scores+softmax+PV
  dim3 agrid(S_LEN / 16, BATCH * NH);
  size_t smbytes = ((size_t)16 * S_LEN + 16 * 8 + 32) * sizeof(float);  // ~129 KB
  attn_kernel<<<agrid, 128, smbytes, stream>>>(qh, kh, vtb, mask, probs, ctx);

  // 4. output projection
  oproj_kernel<<<ptiles / 4, 128, 0, stream>>>(ctx, Wot, bo, out);
}
